// BiRWKVModel_40157944217757
// MI455X (gfx1250) — matrix operand order repaired and checked by reading, not compiled, not run on hardware
//
#include <hip/hip_runtime.h>
#include <hip/hip_bf16.h>
#include <math.h>

// BiRWKV forward for MI455X (gfx1250, wave32).
// HBM-bound model (~620MB weights + 412MB logits @ 23.3TB/s => ~50us floor):
// fp32 weights read once (M-tiles are the fast grid dim so sibling blocks share
// each weight tile through the 192MB L2 -- critical for the 206MB head matrix),
// f16 WMMA with fp32 accumulation for all matmuls, activations/residuals fused
// into GEMM epilogues, nontemporal stores for the write-once logits.

#define L_   4
#define C_   1024
#define F_   4096
#define V_   50304
#define B_   2
#define T_   1024
#define MTOK (B_*T_)   // 2048 rows in every GEMM

typedef __attribute__((ext_vector_type(16))) _Float16 v16h;
typedef __attribute__((ext_vector_type(8)))  _Float16 v8h;
typedef __attribute__((ext_vector_type(8)))  float    v8f;

// ---------------- embedding gather ----------------
__global__ void k_embed(const int* __restrict__ idx, const float* __restrict__ emb,
                        float* __restrict__ x) {
  size_t i = (size_t)blockIdx.x * blockDim.x + threadIdx.x;  // over MTOK*C_
  int tok = (int)(i >> 10);        // / C_
  int c   = (int)(i & (C_ - 1));   // % C_
  x[i] = emb[(size_t)idx[tok] * C_ + c];
}

// ---------------- layernorm (one block per row of C_=1024) ----------------
__global__ void k_layernorm(const float* __restrict__ x, const float* __restrict__ w,
                            const float* __restrict__ b, float* __restrict__ out) {
  __shared__ float sred[256];
  const int tid = threadIdx.x;
  const float* xr = x + (size_t)blockIdx.x * C_;
  float s = 0.f;
  for (int i = tid; i < C_; i += 256) s += xr[i];
  sred[tid] = s; __syncthreads();
  for (int off = 128; off > 0; off >>= 1) {
    if (tid < off) sred[tid] += sred[tid + off];
    __syncthreads();
  }
  const float mu = sred[0] * (1.f / C_);
  __syncthreads();
  float vv = 0.f;
  for (int i = tid; i < C_; i += 256) { float d = xr[i] - mu; vv += d * d; }
  sred[tid] = vv; __syncthreads();
  for (int off = 128; off > 0; off >>= 1) {
    if (tid < off) sred[tid] += sred[tid + off];
    __syncthreads();
  }
  const float rstd = rsqrtf(sred[0] * (1.f / C_) + 1e-5f);
  float* o = out + (size_t)blockIdx.x * C_;
  for (int i = tid; i < C_; i += 256) o[i] = (xr[i] - mu) * rstd * w[i] + b[i];
}

// ---------------- bidirectional WKV scan ----------------
// One thread per (batch, channel): backward exclusive scan to scratch,
// then forward exclusive scan fused with the output ratio.
__global__ void k_wkv(const float* __restrict__ r, const float* __restrict__ k,
                      const float* __restrict__ v, const float* __restrict__ wd,
                      const float* __restrict__ ub, float* __restrict__ numb,
                      float* __restrict__ denb, float* __restrict__ y) {
  int g  = blockIdx.x * blockDim.x + threadIdx.x;   // 0 .. B_*C_-1
  int b  = g >> 10;              // / C_
  int ch = g & (C_ - 1);         // % C_
  const float decay = __expf(-__expf(wd[ch]));
  const float eu    = __expf(ub[ch]);
  const size_t base = (size_t)b * T_ * C_ + ch;
  float akv = 0.f, ak = 0.f;
  for (int t = T_ - 1; t >= 0; --t) {               // backward, exclusive
    size_t i = base + (size_t)t * C_;
    numb[i] = akv; denb[i] = ak;
    float ek = __expf(k[i]);
    akv = akv * decay + ek * v[i];
    ak  = ak  * decay + ek;
  }
  akv = 0.f; ak = 0.f;
  for (int t = 0; t < T_; ++t) {                    // forward, exclusive + output
    size_t i = base + (size_t)t * C_;
    float ek = __expf(k[i]); float vv = v[i];
    float num = akv + numb[i] + eu * vv;
    float den = ak  + denb[i] + eu;
    y[i] = r[i] * num / (den + 1e-8f);
    akv = akv * decay + ek * vv;
    ak  = ak  * decay + ek;
  }
}

// ---------------- WMMA GEMM: Y[M,N] (+)= epi( X[M,K] @ W[K,N] ) ----------------
// Tile 64x64, BK=32, 256 threads = 8 waves in a 4(M) x 2(N) grid;
// each wave owns 16 rows x 32 cols = two v_wmma_f32_16x16x32_f16 accumulators.
// Grid: blockIdx.x = M tile (fast) so blocks sharing a weight tile co-run and
// the weight matrix streams through HBM exactly once (L2 serves the reuse).
// epi: 0=store  1=sigmoid  2=relu^2  3=Y+=  4=Y+=gate*  5=nontemporal store
__global__ __launch_bounds__(256)
void k_gemm_wmma(const float* __restrict__ X, const float* __restrict__ W,
                 float* __restrict__ Y, const float* __restrict__ gate,
                 int M, int N, int K, int epi) {
  constexpr int AS = 40;  // row stride in halves (80B: 16B-aligned, bank-skewed)
  constexpr int BS = 40;
  __shared__ __align__(16) _Float16 Asl[64 * AS];   // A tile, row-major [m][k]
  __shared__ __align__(16) _Float16 Bsl[64 * BS];   // B tile, transposed [n][k]

  const int tid  = threadIdx.x;
  const int lane = tid & 31;
  const int wave = tid >> 5;
  const int wm   = wave & 3;        // M sub-tile 0..3
  const int wn   = wave >> 2;       // N pair 0..1
  const int half = lane >> 4;       // 0|1
  const int mrow = lane & 15;

  const int bm = blockIdx.x * 64;   // M tile: fast grid dim
  const int bn = blockIdx.y * 64;   // N tile: slow grid dim

  // staging map: A -> 8 contiguous k per thread; B -> 8 contiguous n per thread
  const int am  = tid >> 2;         // 0..63
  const int akq = (tid & 3) * 8;    // 0,8,16,24
  const int bkk = tid >> 3;         // 0..31
  const int bnq = (tid & 7) * 8;    // 0..56

  const float* aptr = X + (size_t)(bm + am) * K + akq;
  const float* bptr = W + (size_t)bkk * N + bn + bnq;

  v8f c0 = {}; v8f c1 = {};

  const int ksteps = K >> 5;
  for (int kt = 0; kt < ksteps; ++kt) {
    // ---- stage A (fp32 -> f16) ----
    float4 a0 = *(const float4*)(aptr);
    float4 a1 = *(const float4*)(aptr + 4);
    _Float16* ad = &Asl[am * AS + akq];
    ad[0]=(_Float16)a0.x; ad[1]=(_Float16)a0.y; ad[2]=(_Float16)a0.z; ad[3]=(_Float16)a0.w;
    ad[4]=(_Float16)a1.x; ad[5]=(_Float16)a1.y; ad[6]=(_Float16)a1.z; ad[7]=(_Float16)a1.w;
    // ---- stage B transposed (fp32 -> f16) ----
    float4 b0 = *(const float4*)(bptr);
    float4 b1 = *(const float4*)(bptr + 4);
    _Float16* bd = &Bsl[bnq * BS + bkk];
    bd[0*BS]=(_Float16)b0.x; bd[1*BS]=(_Float16)b0.y; bd[2*BS]=(_Float16)b0.z; bd[3*BS]=(_Float16)b0.w;
    bd[4*BS]=(_Float16)b1.x; bd[5*BS]=(_Float16)b1.y; bd[6*BS]=(_Float16)b1.z; bd[7*BS]=(_Float16)b1.w;

    aptr += 32;
    bptr += (size_t)32 * N;
    if (kt + 1 < ksteps) {                 // gfx1250 global_prefetch_b8
      __builtin_prefetch(aptr, 0, 3);
      __builtin_prefetch(bptr, 0, 3);
    }
    __syncthreads();

    // ---- build fragments in contiguous k order (k = 16*half + s), same for A and B ----
    // A 16x32: lanes 0-15 row m: K0..15; lanes 16-31: K16..31
    union { v16h v; v8h h[2]; } af, bf0, bf1;
    const _Float16* ap = &Asl[(wm * 16 + mrow) * AS];
    af.h[0] = *(const v8h*)(ap + half * 16);
    af.h[1] = *(const v8h*)(ap + half * 16 + 8);
    // B 32x16: lanes 0-15 col n: K0..15; lanes 16-31: K16..31
    const _Float16* bp0 = &Bsl[(wn * 32 + mrow) * BS + half * 16];
    bf0.h[0] = *(const v8h*)(bp0);
    bf0.h[1] = *(const v8h*)(bp0 + 8);
    const _Float16* bp1 = &Bsl[(wn * 32 + 16 + mrow) * BS + half * 16];
    bf1.h[0] = *(const v8h*)(bp1);
    bf1.h[1] = *(const v8h*)(bp1 + 8);

    c0 = __builtin_amdgcn_wmma_f32_16x16x32_f16(false, af.v, false, bf0.v,
                                                (short)0, c0, false, false);
    c1 = __builtin_amdgcn_wmma_f32_16x16x32_f16(false, af.v, false, bf1.v,
                                                (short)0, c1, false, false);
    __syncthreads();
  }

  // ---- fused epilogue; C/D layout: vgpr r -> row r+8*half, col = lane&15 ----
  const int rowBase = bm + wm * 16 + half * 8;
  const int col0    = bn + wn * 32 + mrow;
  #pragma unroll
  for (int vr = 0; vr < 8; ++vr) {
    const size_t rb = (size_t)(rowBase + vr) * N;
    const size_t i0 = rb + col0, i1 = rb + col0 + 16;
    float y0 = c0[vr], y1 = c1[vr];
    if (epi == 0)      { Y[i0] = y0; Y[i1] = y1; }
    else if (epi == 1) { Y[i0] = 1.f / (1.f + __expf(-y0));
                         Y[i1] = 1.f / (1.f + __expf(-y1)); }
    else if (epi == 2) { float t0 = fmaxf(y0, 0.f), t1 = fmaxf(y1, 0.f);
                         Y[i0] = t0 * t0; Y[i1] = t1 * t1; }
    else if (epi == 3) { Y[i0] += y0; Y[i1] += y1; }
    else if (epi == 4) { Y[i0] += gate[i0] * y0; Y[i1] += gate[i1] * y1; }
    else               { __builtin_nontemporal_store(y0, &Y[i0]);   // logits:
                         __builtin_nontemporal_store(y1, &Y[i1]); } // keep L2 for W
  }
}

// ---------------- orchestration ----------------
extern "C" void kernel_launch(void* const* d_in, const int* in_sizes, int n_in,
                              void* d_out, int out_size, void* d_ws, size_t ws_size,
                              hipStream_t stream) {
  (void)in_sizes; (void)n_in; (void)out_size; (void)ws_size;
  const int*   idx   = (const int*)  d_in[0];
  const float* emb   = (const float*)d_in[1];
  const float* ln1w  = (const float*)d_in[2];
  const float* ln1b  = (const float*)d_in[3];
  const float* Wr    = (const float*)d_in[4];
  const float* Wk    = (const float*)d_in[5];
  const float* Wv    = (const float*)d_in[6];
  const float* wd    = (const float*)d_in[7];
  const float* ub    = (const float*)d_in[8];
  const float* Wo    = (const float*)d_in[9];
  const float* ln2w  = (const float*)d_in[10];
  const float* ln2b  = (const float*)d_in[11];
  const float* Wfk   = (const float*)d_in[12];
  const float* Wfv   = (const float*)d_in[13];
  const float* Wfr   = (const float*)d_in[14];
  const float* lnow  = (const float*)d_in[15];
  const float* lnob  = (const float*)d_in[16];
  const float* headW = (const float*)d_in[17];
  float* out = (float*)d_out;

  // workspace: 9*(M*C) + M*F floats = ~108 MB
  const size_t MC = (size_t)MTOK * C_;
  float* ws = (float*)d_ws;
  float* x  = ws;          // residual stream
  float* xn = x  + MC;     // layernorm output
  float* rb = xn + MC;
  float* kb = rb + MC;
  float* vb = kb + MC;
  float* yb = vb + MC;
  float* gt = yb + MC;
  float* nb = gt + MC;     // wkv backward numerator
  float* db = nb + MC;     // wkv backward denominator
  float* kf = db + MC;     // M x F ffn hidden

  const dim3 blk(256);
  const dim3 gCC(MTOK / 64, C_ / 64);   // M fast, N slow => weights stream once
  const dim3 gCF(MTOK / 64, F_ / 64);
  const dim3 gV (MTOK / 64, V_ / 64);   // 50304 = 64 * 786

  k_embed<<<dim3((unsigned)(MC / 256)), blk, 0, stream>>>(idx, emb, x);

  for (int l = 0; l < L_; ++l) {
    const float* wr  = Wr  + (size_t)l * C_ * C_;
    const float* wk  = Wk  + (size_t)l * C_ * C_;
    const float* wv  = Wv  + (size_t)l * C_ * C_;
    const float* wo  = Wo  + (size_t)l * C_ * C_;
    const float* wfk = Wfk + (size_t)l * C_ * F_;
    const float* wfv = Wfv + (size_t)l * F_ * C_;
    const float* wfr = Wfr + (size_t)l * C_ * C_;

    k_layernorm<<<MTOK, blk, 0, stream>>>(x, ln1w + l * C_, ln1b + l * C_, xn);
    k_gemm_wmma<<<gCC, blk, 0, stream>>>(xn, wr, rb, nullptr, MTOK, C_, C_, 1); // sigmoid
    k_gemm_wmma<<<gCC, blk, 0, stream>>>(xn, wk, kb, nullptr, MTOK, C_, C_, 0);
    k_gemm_wmma<<<gCC, blk, 0, stream>>>(xn, wv, vb, nullptr, MTOK, C_, C_, 0);
    k_wkv<<<dim3((B_ * C_) / 256), blk, 0, stream>>>(rb, kb, vb, wd + l * C_,
                                                     ub + l * C_, nb, db, yb);
    k_gemm_wmma<<<gCC, blk, 0, stream>>>(yb, wo, x, nullptr, MTOK, C_, C_, 3);  // x +=
    k_layernorm<<<MTOK, blk, 0, stream>>>(x, ln2w + l * C_, ln2b + l * C_, xn);
    k_gemm_wmma<<<gCF, blk, 0, stream>>>(xn, wfk, kf, nullptr, MTOK, F_, C_, 2); // relu^2
    k_gemm_wmma<<<gCC, blk, 0, stream>>>(xn, wfr, gt, nullptr, MTOK, C_, C_, 1); // sigmoid
    k_gemm_wmma<<<gCC, blk, 0, stream>>>(kf, wfv, x, gt, MTOK, C_, F_, 4);       // x += g*
  }

  k_layernorm<<<MTOK, blk, 0, stream>>>(x, lnow, lnob, xn);
  k_gemm_wmma<<<gV, blk, 0, stream>>>(xn, headW, out, nullptr, MTOK, V_, C_, 5); // NT store
}
